// SMPL_33028298506333
// MI455X (gfx1250) — matrix operand order repaired and checked by reading, not compiled, not run on hardware
//
#include <hip/hip_runtime.h>

// ---------------------------------------------------------------------------
// SMPL forward on gfx1250 (wave32, WMMA bf16 16x16x32, fp32 accumulate)
//
// Pipeline:
//   k_jconst : JS = J_regressor @ shapedirs (24x3x10), Jt = J_regressor @ v_template
//   k_pack   : bf16-pack D = [shapedirs|posedirs] column-major (Bcm[col][224]),
//              weights -> wpack[v][32], v_template -> padded f32
//   k_pose   : per-n rodrigues, lrotmin coefs (bf16 A rows), joints J,
//              kinematic chain -> G_rel top 3 rows packed as WMMA B frag (gpack)
//   k_gemm1  : v_posed = coef[2048x224] x Bcm[224x20672]  (WMMA bf16, K=7x32)
//   k_skin   : per (n, 16-vertex tile): T = weights x G_rel via one WMMA,
//              LDS epilogue applies 3x4 affine + trans -> verts
//   k_joints : joints = joint_regressor^T @ verts (block per n, shfl+LDS reduce)
// ---------------------------------------------------------------------------

typedef __bf16 bf16;
typedef __attribute__((ext_vector_type(16))) __bf16        v16bf;
typedef __attribute__((ext_vector_type(8)))  float         v8f;
typedef __attribute__((ext_vector_type(4)))  unsigned int  u32x4;

union FragU { u32x4 q[2]; v16bf v; };   // 32 bytes = one 16-elem bf16 operand/lane

#define SMPL_V     6890
#define SMPL_NJ    24
#define SMPL_NB    10
#define SMPL_NP    207
#define SMPL_NK    19
#define K1         224          // 217 padded to 7*32
#define KB1        7
#define COLS       (SMPL_V*3)   // 20670
#define COLSP      20672        // padded to 16
#define NCT        (COLSP/16)   // 1292 column tiles
#define VPAD       6912         // vertex rows padded to 16 (431*16 = 6896 <= 6912)
#define VTILES     431          // ceil(6890/16)

// ---------------------------------------------------------------------------
// Kernel 0: joint-regression constants. jconst[j][0..29]=JS[c*10+b], [30..32]=Jt[c]
__global__ void k_jconst(const float* __restrict__ jreg,
                         const float* __restrict__ sdirs,
                         const float* __restrict__ vtemp,
                         float* __restrict__ jconst) {
  const int j = blockIdx.x;
  const int t = threadIdx.x;              // 0..32
  const float* row = jreg + (size_t)j * SMPL_V;
  float acc = 0.f;
  if (t < 30) {
    const int c = t / 10, b = t % 10;
    for (int v = 0; v < SMPL_V; ++v) acc += row[v] * sdirs[v * 30 + c * 10 + b];
  } else {
    const int c = t - 30;
    for (int v = 0; v < SMPL_V; ++v) acc += row[v] * vtemp[v * 3 + c];
  }
  jconst[j * 33 + t] = acc;
}

// ---------------------------------------------------------------------------
// Kernel 1: bf16 packing of the static matrices.
__device__ __forceinline__ float dirval(const float* sdirs, const float* pdirs,
                                        int v, int c, int k) {
  if (k < SMPL_NB)  return sdirs[v * 30 + c * 10 + k];
  if (k < 217)      return pdirs[v * 621 + c * 207 + (k - SMPL_NB)];
  return 0.f;
}

__global__ __launch_bounds__(256) void k_pack(const float* __restrict__ sdirs,
                                              const float* __restrict__ pdirs,
                                              const float* __restrict__ vtemp,
                                              const float* __restrict__ weights,
                                              bf16* __restrict__ bcm,
                                              bf16* __restrict__ wpack,
                                              float* __restrict__ vtpad) {
  const long i = (long)blockIdx.x * 256 + threadIdx.x;
  const long R1 = (long)COLSP * (K1 / 2);       // Bcm dword-pairs
  const long R2 = (long)VPAD * 16;              // wpack dword-pairs
  const long R3 = COLSP;                        // vtpad
  if (i < R1) {
    const int col = (int)(i / (K1 / 2));
    const int kd  = (int)(i % (K1 / 2));
    float f0 = 0.f, f1 = 0.f;
    if (col < COLS) {
      const int v = col / 3, c = col % 3, k0 = kd * 2;
      f0 = dirval(sdirs, pdirs, v, c, k0);
      f1 = dirval(sdirs, pdirs, v, c, k0 + 1);
    }
    bcm[(size_t)col * K1 + kd * 2]     = (bf16)f0;
    bcm[(size_t)col * K1 + kd * 2 + 1] = (bf16)f1;
  } else if (i < R1 + R2) {
    const long p = i - R1;
    const int v = (int)(p / 16), kd = (int)(p % 16), k0 = kd * 2;
    float f0 = (v < SMPL_V && k0     < SMPL_NJ) ? weights[v * SMPL_NJ + k0]     : 0.f;
    float f1 = (v < SMPL_V && k0 + 1 < SMPL_NJ) ? weights[v * SMPL_NJ + k0 + 1] : 0.f;
    wpack[(size_t)v * 32 + k0]     = (bf16)f0;
    wpack[(size_t)v * 32 + k0 + 1] = (bf16)f1;
  } else if (i < R1 + R2 + R3) {
    const int col = (int)(i - R1 - R2);
    vtpad[col] = (col < COLS) ? vtemp[col] : 0.f;
  }
}

// ---------------------------------------------------------------------------
// Kernel 2: per-batch pose prep (rodrigues, coefs, joints, kinematic chain).
__global__ void k_pose(const float* __restrict__ beta,
                       const float* __restrict__ theta,
                       const float* __restrict__ jconst,
                       bf16* __restrict__ coef,
                       bf16* __restrict__ gpack,
                       int nbatch) {
  const int n = blockIdx.x * blockDim.x + threadIdx.x;
  if (n >= nbatch) return;

  float R[SMPL_NJ][9];
  const float* th = theta + (size_t)n * SMPL_NJ * 3;
  for (int j = 0; j < SMPL_NJ; ++j) {
    const float x = th[j * 3], y = th[j * 3 + 1], z = th[j * 3 + 2];
    float t = sqrtf(x * x + y * y + z * z);
    t = fmaxf(t, 1.17549435e-38f);
    const float inv = 1.0f / t;
    const float ux = x * inv, uy = y * inv, uz = z * inv;
    const float c = cosf(t), s = sinf(t), oc = 1.0f - c;
    R[j][0] = c + oc * ux * ux;  R[j][1] = oc * ux * uy - s * uz;  R[j][2] = oc * ux * uz + s * uy;
    R[j][3] = oc * uy * ux + s * uz;  R[j][4] = c + oc * uy * uy;  R[j][5] = oc * uy * uz - s * ux;
    R[j][6] = oc * uz * ux - s * uy;  R[j][7] = oc * uz * uy + s * ux;  R[j][8] = c + oc * uz * uz;
  }

  // A row: [beta(10) | lrotmin(207) | pad] as bf16
  bf16* cr = coef + (size_t)n * K1;
  const float* b = beta + (size_t)n * SMPL_NB;
  for (int k = 0; k < SMPL_NB; ++k) cr[k] = (bf16)b[k];
  for (int j = 1; j < SMPL_NJ; ++j)
    for (int idx = 0; idx < 9; ++idx) {
      const float e = (idx == 0 || idx == 4 || idx == 8) ? 1.f : 0.f;
      cr[SMPL_NB + (j - 1) * 9 + idx] = (bf16)(R[j][idx] - e);
    }
  for (int k = 217; k < K1; ++k) cr[k] = (bf16)0.f;

  // Joints from precomputed constants
  float J[SMPL_NJ][3];
  for (int j = 0; j < SMPL_NJ; ++j) {
    const float* jc = jconst + j * 33;
    for (int c = 0; c < 3; ++c) {
      float acc = jc[30 + c];
      for (int q = 0; q < SMPL_NB; ++q) acc += jc[c * 10 + q] * b[q];
      J[j][c] = acc;
    }
  }

  // Kinematic chain: affine top 3 rows [R|t] per joint
  const int par[SMPL_NJ] = {-1,0,0,0,1,2,3,4,5,6,7,8,9,9,9,12,13,14,16,17,18,19,20,21};
  float G[SMPL_NJ][12];
  for (int r = 0; r < 3; ++r) {
    G[0][r * 4 + 0] = R[0][r * 3 + 0];
    G[0][r * 4 + 1] = R[0][r * 3 + 1];
    G[0][r * 4 + 2] = R[0][r * 3 + 2];
    G[0][r * 4 + 3] = J[0][r];
  }
  for (int j = 1; j < SMPL_NJ; ++j) {
    const int p = par[j];
    const float tj0 = J[j][0] - J[p][0];
    const float tj1 = J[j][1] - J[p][1];
    const float tj2 = J[j][2] - J[p][2];
    for (int r = 0; r < 3; ++r) {
      const float g0 = G[p][r * 4 + 0], g1 = G[p][r * 4 + 1];
      const float g2 = G[p][r * 4 + 2], g3 = G[p][r * 4 + 3];
      for (int cc = 0; cc < 3; ++cc)
        G[j][r * 4 + cc] = g0 * R[j][0 * 3 + cc] + g1 * R[j][1 * 3 + cc] + g2 * R[j][2 * 3 + cc];
      G[j][r * 4 + 3] = g0 * tj0 + g1 * tj1 + g2 * tj2 + g3;
    }
  }
  // subtract rest-pose joint contribution (translation column only)
  for (int j = 0; j < SMPL_NJ; ++j)
    for (int r = 0; r < 3; ++r)
      G[j][r * 4 + 3] -= G[j][r * 4 + 0] * J[j][0] + G[j][r * 4 + 1] * J[j][1] +
                         G[j][r * 4 + 2] * J[j][2];

  // Pack as WMMA B fragment (col a = lane%16, K j = e + 16*(lane/16))
  bf16* gp = gpack + (size_t)n * 512;
  for (int l = 0; l < 32; ++l)
    for (int e = 0; e < 16; ++e) {
      const int j = e + ((l < 16) ? 0 : 16);
      const int a = l & 15;
      const float val = (j < SMPL_NJ && a < 12) ? G[j][a] : 0.f;
      gp[l * 16 + e] = (bf16)val;
    }
}

// ---------------------------------------------------------------------------
// Kernel 3: blendshape GEMM. C[16n x 16col] tiles, K = 7 x 32 bf16 WMMA.
__global__ __launch_bounds__(256) void k_gemm1(const unsigned int* __restrict__ coef, // dwords
                                               const bf16* __restrict__ bcm,
                                               const float* __restrict__ vtpad,
                                               float* __restrict__ vpos) {
  __shared__ unsigned int lA[16 * (K1 / 2)];          // 16 rows x 224 bf16 = 7 KB
  const int nt = blockIdx.y;
  const unsigned int* src = coef + (size_t)nt * 16 * (K1 / 2);
  for (int d = threadIdx.x; d < 16 * (K1 / 2); d += 256) lA[d] = src[d];
  __syncthreads();

  const int wave = threadIdx.x >> 5;
  const int lane = threadIdx.x & 31;

  // Build 7 A fragments from LDS (ISA A-layout: two 16B chunks per lane per kb)
  FragU a[KB1];
  {
    const char* base = (const char*)lA + (lane & 15) * (K1 * 2) + ((lane < 16) ? 0 : 16);
    #pragma unroll
    for (int kb = 0; kb < KB1; ++kb) {
      a[kb].q[0] = *(const u32x4*)(base + kb * 64);
      a[kb].q[1] = *(const u32x4*)(base + kb * 64 + 32);
    }
  }

  const int ct = blockIdx.x * 8 + wave;
  if (ct >= NCT) return;                               // wave-uniform exit

  const int col = ct * 16 + (lane & 15);
  const char* bbase = (const char*)bcm + (size_t)col * (K1 * 2) + ((lane < 16) ? 0 : 16);
  v8f acc = {};
  #pragma unroll
  for (int kb = 0; kb < KB1; ++kb) {
    FragU bf;
    bf.q[0] = *(const u32x4*)(bbase + kb * 64);
    bf.q[1] = *(const u32x4*)(bbase + kb * 64 + 32);
    acc = __builtin_amdgcn_wmma_f32_16x16x32_bf16(false, a[kb].v, false, bf.v,
                                                  (short)0, acc, false, false);
  }
  const float tv = vtpad[col];
  const int rbase = nt * 16 + ((lane < 16) ? 0 : 8);
  #pragma unroll
  for (int r = 0; r < 8; ++r)
    vpos[(size_t)(rbase + r) * COLSP + col] = acc[r] + tv;
}

// ---------------------------------------------------------------------------
// Kernel 4: skinning. One WMMA (weights[16v x 24] x Gtop[24 x 12]) per tile,
// LDS epilogue applies the affine transform + trans.
__global__ __launch_bounds__(128) void k_skin(const bf16* __restrict__ wpack,
                                              const bf16* __restrict__ gpack,
                                              const float* __restrict__ vpos,
                                              const float* __restrict__ trans,
                                              float* __restrict__ verts) {
  __shared__ float tc[4][16 * 17];                     // per-wave C tile, padded
  const int n    = blockIdx.x;
  const int wave = threadIdx.x >> 5;
  const int lane = threadIdx.x & 31;
  const int g    = blockIdx.y * 4 + wave;              // 0..15 wave slot

  FragU bfr;                                           // G_rel fragment (reused)
  {
    const char* gb = (const char*)gpack + (size_t)n * 1024 + lane * 32;
    bfr.q[0] = ((const u32x4*)gb)[0];
    bfr.q[1] = ((const u32x4*)gb)[1];
  }
  const float t0 = trans[n * 3 + 0], t1 = trans[n * 3 + 1], t2 = trans[n * 3 + 2];

  for (int i = 0; i < 27; ++i) {                       // uniform trip count
    const int vt = g + i * 16;
    if (vt < VTILES) {
      const int v = vt * 16 + (lane & 15);
      const char* ab = (const char*)wpack + (size_t)v * 64 + ((lane < 16) ? 0 : 32);
      FragU afr;
      afr.q[0] = *(const u32x4*)ab;
      afr.q[1] = *(const u32x4*)(ab + 16);
      v8f acc = {};
      acc = __builtin_amdgcn_wmma_f32_16x16x32_bf16(false, afr.v, false, bfr.v,
                                                    (short)0, acc, false, false);
      const int m0 = (lane < 16) ? 0 : 8;
      #pragma unroll
      for (int r = 0; r < 8; ++r) tc[wave][(m0 + r) * 17 + (lane & 15)] = acc[r];
    }
    __syncthreads();
    if (vt < VTILES && lane < 16) {
      const int v = vt * 16 + lane;
      if (v < SMPL_V) {
        const float* T = &tc[wave][lane * 17];
        const float p0 = vpos[(size_t)n * COLSP + v * 3 + 0];
        const float p1 = vpos[(size_t)n * COLSP + v * 3 + 1];
        const float p2 = vpos[(size_t)n * COLSP + v * 3 + 2];
        float* dst = verts + ((size_t)n * SMPL_V + v) * 3;
        dst[0] = T[0] * p0 + T[1] * p1 + T[2]  * p2 + T[3]  + t0;
        dst[1] = T[4] * p0 + T[5] * p1 + T[6]  * p2 + T[7]  + t1;
        dst[2] = T[8] * p0 + T[9] * p1 + T[10] * p2 + T[11] + t2;
      }
    }
    __syncthreads();
  }
}

// ---------------------------------------------------------------------------
// Kernel 5: joint regression over verts. Block per n, shuffle+LDS reduction.
__global__ __launch_bounds__(256) void k_joints(const float* __restrict__ verts,
                                                const float* __restrict__ jr,
                                                float* __restrict__ jout) {
  const int n = blockIdx.x, tid = threadIdx.x;
  float acc[SMPL_NK * 3];
  #pragma unroll
  for (int o = 0; o < SMPL_NK * 3; ++o) acc[o] = 0.f;
  const float* vb = verts + (size_t)n * SMPL_V * 3;
  for (int v = tid; v < SMPL_V; v += 256) {
    const float x = vb[v * 3], y = vb[v * 3 + 1], z = vb[v * 3 + 2];
    const float* w = jr + (size_t)v * SMPL_NK;
    #pragma unroll
    for (int k = 0; k < SMPL_NK; ++k) {
      const float ww = w[k];
      acc[k * 3 + 0] += ww * x;
      acc[k * 3 + 1] += ww * y;
      acc[k * 3 + 2] += ww * z;
    }
  }
  #pragma unroll
  for (int o = 0; o < SMPL_NK * 3; ++o)
    for (int off = 16; off > 0; off >>= 1)
      acc[o] += __shfl_down(acc[o], off, 32);
  __shared__ float red[8][SMPL_NK * 3];
  const int w = tid >> 5, lane = tid & 31;
  if (lane == 0)
    for (int o = 0; o < SMPL_NK * 3; ++o) red[w][o] = acc[o];
  __syncthreads();
  if (tid < SMPL_NK * 3) {
    float s = 0.f;
    for (int q = 0; q < 8; ++q) s += red[q][tid];
    jout[(size_t)n * SMPL_NK * 3 + tid] = s;
  }
}

// ---------------------------------------------------------------------------
extern "C" void kernel_launch(void* const* d_in, const int* in_sizes, int n_in,
                              void* d_out, int out_size, void* d_ws, size_t ws_size,
                              hipStream_t stream) {
  const float* beta    = (const float*)d_in[0];
  const float* theta   = (const float*)d_in[1];
  const float* trans   = (const float*)d_in[2];
  const float* sdirs   = (const float*)d_in[3];
  const float* vtemp   = (const float*)d_in[4];
  const float* jreg    = (const float*)d_in[5];
  const float* pdirs   = (const float*)d_in[6];
  const float* weights = (const float*)d_in[7];
  const float* jregK   = (const float*)d_in[8];
  const int nb = in_sizes[0] / SMPL_NB;     // 2048 (multiple of 16)

  char* ws = (char*)d_ws;
  size_t off = 0;
  auto carve = [&](size_t bytes) { char* p = ws + off; off = (off + bytes + 255) & ~(size_t)255; return p; };
  bf16*  bcm    = (bf16*)carve((size_t)COLSP * K1 * 2);      // 9.26 MB
  bf16*  coef   = (bf16*)carve((size_t)nb * K1 * 2);         // 0.92 MB
  bf16*  wpack  = (bf16*)carve((size_t)VPAD * 32 * 2);       // 0.44 MB
  float* vtpad  = (float*)carve((size_t)COLSP * 4);          // 83 KB
  float* jconst = (float*)carve((size_t)SMPL_NJ * 33 * 4);   // 3 KB
  bf16*  gpack  = (bf16*)carve((size_t)nb * 512 * 2);        // 2.1 MB
  float* vpos   = (float*)carve((size_t)nb * COLSP * 4);     // 169 MB

  float* verts = (float*)d_out;
  float* jout  = verts + (size_t)nb * SMPL_V * 3;

  k_jconst<<<SMPL_NJ, 33, 0, stream>>>(jreg, sdirs, vtemp, jconst);

  const long packN = (long)COLSP * (K1 / 2) + (long)VPAD * 16 + COLSP;
  k_pack<<<(int)((packN + 255) / 256), 256, 0, stream>>>(sdirs, pdirs, vtemp, weights,
                                                         bcm, wpack, vtpad);

  k_pose<<<(nb + 255) / 256, 256, 0, stream>>>(beta, theta, jconst, coef, gpack, nb);

  k_gemm1<<<dim3((NCT + 7) / 8, nb / 16), 256, 0, stream>>>((const unsigned int*)coef,
                                                            bcm, vtpad, vpos);

  k_skin<<<dim3(nb, 4), 128, 0, stream>>>(wpack, gpack, vpos, trans, verts);

  k_joints<<<nb, 256, 0, stream>>>(verts, jregK, jout);
}
